// PrimaryNet_33114197852630
// MI455X (gfx1250) — matrix operand order repaired and checked by reading, not compiled, not run on hardware
//
#include <hip/hip_runtime.h>
#include <stdint.h>

typedef _Float16 h16;
typedef __attribute__((ext_vector_type(16))) _Float16 v16h;
typedef __attribute__((ext_vector_type(8)))  float    v8f;

#define B_ 8
#define H_ 256
#define W_ 256
#define C_ 64
#define HW_ (H_*W_)
#define EPS_ 1e-5f
#define SLOPE_ 0.01f
#define NCROP_ 244
#define NPIX_ (NCROP_*NCROP_)

#if defined(__has_builtin)
#if __has_builtin(__builtin_amdgcn_global_load_async_to_lds_b128)
#define HAS_ASYNC_LDS 1
#endif
#endif

#ifdef HAS_ASYNC_LDS
typedef int v4i_ __attribute__((vector_size(16)));           // matches builtin's V4i
typedef __attribute__((address_space(1))) v4i_* as1_v4i;     // global
typedef __attribute__((address_space(3))) v4i_* as3_v4i;     // LDS
#endif

__device__ __forceinline__ int clampi(int v, int lo, int hi) {
  return v < lo ? lo : (v > hi ? hi : v);
}

__device__ __forceinline__ void wait_async_then_barrier() {
#ifdef HAS_ASYNC_LDS
#if __has_builtin(__builtin_amdgcn_s_wait_asynccnt)
  __builtin_amdgcn_s_wait_asynccnt(0);
#else
  asm volatile("s_wait_asynccnt 0x0" ::: "memory");
#endif
#endif
  __syncthreads();
}

// ---------------------------------------------------------------------------
// Weight pre-transform: ws (7,64,64,3,3) f32 -> wT[l][tap][cout][cin] f16
// ---------------------------------------------------------------------------
__global__ void k_wprep(const float* __restrict__ ws, h16* __restrict__ wT) {
  int idx = blockIdx.x * 256 + threadIdx.x;
  if (idx >= 7 * 9 * 64 * 64) return;
  int l   = idx / 36864;
  int r   = idx % 36864;
  int tap = r / 4096;
  int o   = (r >> 6) & 63;
  int i   = r & 63;
  float v = ws[(size_t)(((l * 64 + o) * 64 + i) * 9) + tap];
  wT[idx] = (h16)v;
}

// ---------------------------------------------------------------------------
// First conv: 1 -> 64, replication pad, raw f32 channels-last + stats
// ---------------------------------------------------------------------------
__global__ void __launch_bounds__(256)
k_conv0(const float* __restrict__ xIn, const float* __restrict__ w0,
        const float* __restrict__ b0, float* __restrict__ raw,
        float* __restrict__ stats) {
  __shared__ float sstat[128];
  int tid = threadIdx.x;
  if (tid < 128) sstat[tid] = 0.f;
  __syncthreads();

  int gid = blockIdx.x * 256 + tid;
  int b = gid >> 16;          // HW = 65536
  int p = gid & 65535;
  int y = p >> 8, x = p & 255;

  float v[9];
#pragma unroll
  for (int t = 0; t < 9; ++t) {
    int yc = clampi(y + t / 3 - 1, 0, 255);
    int xc = clampi(x + t % 3 - 1, 0, 255);
    v[t] = xIn[b * HW_ + yc * 256 + xc];
  }

  int lane = tid & 31;
  float* out = raw + (size_t)gid * 64;
  for (int co = 0; co < 64; ++co) {
    float acc = b0[co];
#pragma unroll
    for (int t = 0; t < 9; ++t) acc += w0[co * 9 + t] * v[t];
    out[co] = acc;
    float s1 = acc, s2 = acc * acc;
    for (int off = 16; off; off >>= 1) {
      s1 += __shfl_down(s1, off, 32);
      s2 += __shfl_down(s2, off, 32);
    }
    if (lane == 0) {
      atomicAdd(&sstat[2 * co], s1);
      atomicAdd(&sstat[2 * co + 1], s2);
    }
  }
  __syncthreads();
  if (tid < 128) atomicAdd(&stats[b * 128 + tid], sstat[tid]);
}

// ---------------------------------------------------------------------------
// Stats finalize: mean + rsqrt(var+eps) per (b, c)
// ---------------------------------------------------------------------------
__global__ void k_finalize(const float* __restrict__ stats, float* __restrict__ mrs) {
  int i = blockIdx.x * 256 + threadIdx.x;
  if (i >= B_ * C_) return;
  float n = (float)HW_;
  float m = stats[2 * i] / n;
  float var = stats[2 * i + 1] / n - m * m;
  mrs[2 * i]     = m;
  mrs[2 * i + 1] = rsqrtf(var + EPS_);
}

// ---------------------------------------------------------------------------
// InstanceNorm + LeakyReLU, raw f32 -> act f16 (channels-last)
// ---------------------------------------------------------------------------
__global__ void __launch_bounds__(256)
k_norm(const float* __restrict__ raw, const float* __restrict__ mrs,
       h16* __restrict__ act) {
  size_t idx = (size_t)blockIdx.x * 256 + threadIdx.x;
  int c = (int)(idx & 63);
  int b = (int)(idx >> 22);   // HW*C = 2^22
  float m  = mrs[(b * 64 + c) * 2];
  float rs = mrs[(b * 64 + c) * 2 + 1];
  float xv = (raw[idx] - m) * rs;
  xv = xv >= 0.f ? xv : SLOPE_ * xv;
  act[idx] = (h16)xv;
}

// ---------------------------------------------------------------------------
// Mid conv layers 64->64 via WMMA implicit GEMM, register-blocked over N.
// Block = 256 thr (8 waves): tile = 8 rows x 16 cols x 64 couts.
// Wave w: M-tile = row y0+w (16 pixels), N = all 64 couts (4 accumulators).
// K = tap*64 + cin (576), chunks of 32 -> 18 chunks x 4 WMMA = 72 WMMA/wave.
// Staging uses global_load_async_to_lds_b128 (ASYNCcnt) when available.
// ---------------------------------------------------------------------------
__global__ void __launch_bounds__(256)
k_conv(const h16* __restrict__ act, const h16* __restrict__ wL,
       const float* __restrict__ bias, float* __restrict__ raw,
       float* __restrict__ stats) {
  extern __shared__ __align__(16) char smem[];
  uint32_t* sW   = (uint32_t*)smem;          // [9][64][32] u32: weight f16 pairs
  uint32_t* sIn  = sW + 9 * 64 * 32;         // [10][18][32] u32: halo tile pairs
  float*    sstat = (float*)(sIn + 10 * 18 * 32); // [128]

  int tid = threadIdx.x;
  int x0 = blockIdx.x * 16, y0 = blockIdx.y * 8, b = blockIdx.z;

  if (tid < 128) sstat[tid] = 0.f;

  // ---- Stage full layer weights ([tap][cout][cin] f16) into LDS: 72 KB ----
  const uint4* wsrc = (const uint4*)wL;
  uint4* wdst = (uint4*)sW;
#ifdef HAS_ASYNC_LDS
  for (int i = tid; i < 4608; i += 256)
    __builtin_amdgcn_global_load_async_to_lds_b128(
        (as1_v4i)(wsrc + i), (as3_v4i)(wdst + i), 0, 0);
#else
  for (int i = tid; i < 4608; i += 256) wdst[i] = wsrc[i];
#endif

  // ---- Stage input halo tile (10 rows x 18 cols x 64 ch), replication pad --
  const uint4* asrc = (const uint4*)act;
  uint4* idst = (uint4*)sIn;
  for (int u = tid; u < 1440; u += 256) {
    int slot = u >> 3, sub = u & 7;
    int ry = slot / 18, rx = slot % 18;
    int yc = clampi(y0 - 1 + ry, 0, 255);
    int xc = clampi(x0 - 1 + rx, 0, 255);
    const uint4* gp = asrc + (size_t)(b * HW_ + yc * 256 + xc) * 8 + sub;
#ifdef HAS_ASYNC_LDS
    __builtin_amdgcn_global_load_async_to_lds_b128(
        (as1_v4i)gp, (as3_v4i)(idst + u), 0, 0);
#else
    idst[u] = *gp;
#endif
  }
  wait_async_then_barrier();

  int lane = tid & 31, wave = tid >> 5;
  int hi = lane >> 4, ln = lane & 15;

  v8f acc[4];
#pragma unroll
  for (int g = 0; g < 4; ++g)
    acc[g] = (v8f){0.f, 0.f, 0.f, 0.f, 0.f, 0.f, 0.f, 0.f};

#pragma unroll
  for (int t = 0; t < 18; ++t) {
    int tap = t >> 1, cin0 = (t & 1) * 32;
    int dy = tap / 3 - 1, dx = tap % 3 - 1;
    const uint32_t* pin = &sIn[((wave + 1 + dy) * 18 + (ln + 1 + dx)) * 32];
    union F { v16h h; uint32_t u[8]; } A;
#pragma unroll
    for (int v = 0; v < 8; ++v) {
      // A 16x32 f16 layout (ISA 7.12.2): K = (v>>2)*16 + (hi?8:0) + (v&3)*2
      int ka = ((v >> 2) << 4) + (hi ? 8 : 0) + ((v & 3) << 1);
      A.u[v] = pin[(cin0 + ka) >> 1];
    }
#pragma unroll
    for (int g = 0; g < 4; ++g) {
      const uint32_t* pw = &sW[(tap * 64 + g * 16 + ln) * 32];
      union F Bf;
#pragma unroll
      for (int v = 0; v < 8; ++v) {
        // B 32x16 f16 layout (same K order as A): K = (v>>2)*16 + (hi?8:0) + (v&3)*2
        int kb = ((v >> 2) << 4) + (hi ? 8 : 0) + ((v & 3) << 1);
        Bf.u[v] = pw[(cin0 + kb) >> 1];
      }
      acc[g] = __builtin_amdgcn_wmma_f32_16x16x32_f16(false, A.h, false, Bf.h,
                                                      (short)0, acc[g], false, false);
    }
  }

  // Epilogue: bias, store raw f32 channels-last, fused instance-norm stats.
  size_t base = ((size_t)b * HW_ + (size_t)(y0 + wave) * 256 + x0) * 64;
#pragma unroll
  for (int g = 0; g < 4; ++g) {
    int cout = g * 16 + ln;
    float bv = bias[cout];
    float s1 = 0.f, s2 = 0.f;
#pragma unroll
    for (int v = 0; v < 8; ++v) {
      float d = acc[g][v] + bv;       // D layout: lane=N(cout), VGPR v -> M half
      int m = v + hi * 8;             // pixel x = x0 + m
      raw[base + (size_t)m * 64 + cout] = d;
      s1 += d;
      s2 += d * d;
    }
    s1 += __shfl_xor(s1, 16, 32);     // combine M halves (same cout)
    s2 += __shfl_xor(s2, 16, 32);
    if (!hi) {
      atomicAdd(&sstat[2 * cout], s1);
      atomicAdd(&sstat[2 * cout + 1], s2);
    }
  }
  __syncthreads();
  if (tid < 128) atomicAdd(&stats[b * 128 + tid], sstat[tid]);
}

// ---------------------------------------------------------------------------
// Projection conv 64->1 + residual
// ---------------------------------------------------------------------------
__global__ void __launch_bounds__(256)
k_proj(const h16* __restrict__ act, const float* __restrict__ wp,
       const float* __restrict__ bp, const float* __restrict__ xIn,
       float* __restrict__ hr) {
  int gid = blockIdx.x * 256 + threadIdx.x;
  int b = gid >> 16, p = gid & 65535;
  int y = p >> 8, x = p & 255;
  float acc = bp[0];
#pragma unroll
  for (int t = 0; t < 9; ++t) {
    int yc = clampi(y + t / 3 - 1, 0, 255);
    int xc = clampi(x + t % 3 - 1, 0, 255);
    const h16* pa = act + (size_t)(b * HW_ + yc * 256 + xc) * 64;
    for (int c = 0; c < 64; ++c) acc += (float)pa[c] * wp[c * 9 + t];
  }
  hr[gid] = acc + xIn[gid];
}

// ---------------------------------------------------------------------------
// Shift-search loss: per (shift s, batch b) accumulate S1=sum(d), S2=sum(d^2)
// with d = crop - t; loss_sb = S2/n - (S1/n)^2 (equalizer folds into mean).
// ---------------------------------------------------------------------------
__global__ void __launch_bounds__(256)
k_lossacc(const float* __restrict__ hr, const float* __restrict__ tgt,
          float* __restrict__ lb) {
  __shared__ float red[16];
  int tid = threadIdx.x;
  int pid = blockIdx.x * 256 + tid;
  int bI = blockIdx.y, s = blockIdx.z;
  int sy = s / 13, sx = s % 13;
  float d = 0.f;
  if (pid < NPIX_) {
    int hy = pid / NCROP_, hx = pid % NCROP_;
    float c = hr[bI * HW_ + (sy + hy) * 256 + (sx + hx)];
    float t = tgt[bI * HW_ + (6 + hy) * 256 + (6 + hx)];
    d = c - t;
  }
  float s1 = d, s2 = d * d;
  for (int off = 16; off; off >>= 1) {
    s1 += __shfl_down(s1, off, 32);
    s2 += __shfl_down(s2, off, 32);
  }
  int lane = tid & 31, wv = tid >> 5;
  if (lane == 0) { red[wv] = s1; red[8 + wv] = s2; }
  __syncthreads();
  if (tid == 0) {
    float t1 = 0.f, t2 = 0.f;
    for (int i = 0; i < 8; ++i) { t1 += red[i]; t2 += red[8 + i]; }
    atomicAdd(&lb[(s * 8 + bI) * 2], t1);
    atomicAdd(&lb[(s * 8 + bI) * 2 + 1], t2);
  }
}

__global__ void k_lossfin(const float* __restrict__ lb, float* __restrict__ out) {
  __shared__ float best[8];
  int tid = threadIdx.x;
  if (tid < 8) {
    float bmin = 3.4e38f;
    float n = (float)NPIX_;
    for (int s = 0; s < 169; ++s) {
      float S1 = lb[(s * 8 + tid) * 2];
      float S2 = lb[(s * 8 + tid) * 2 + 1];
      float e = S1 / n;
      float l = S2 / n - e * e;
      bmin = fminf(bmin, l);
    }
    best[tid] = bmin;
  }
  __syncthreads();
  if (tid == 0) {
    float acc = 0.f;
    for (int i = 0; i < 8; ++i) acc += best[i];
    out[0] = acc / 8.f;
  }
}

// ---------------------------------------------------------------------------
extern "C" void kernel_launch(void* const* d_in, const int* in_sizes, int n_in,
                              void* d_out, int out_size, void* d_ws, size_t ws_size,
                              hipStream_t stream) {
  const float* xIn = (const float*)d_in[0];
  const float* tgt = (const float*)d_in[1];
  const float* w0  = (const float*)d_in[2];
  const float* b0  = (const float*)d_in[3];
  const float* wsA = (const float*)d_in[4];
  const float* bsA = (const float*)d_in[5];
  const float* wp  = (const float*)d_in[6];
  const float* bp  = (const float*)d_in[7];

  char* ws = (char*)d_ws;
  size_t off = 0;
  auto alloc = [&](size_t bytes) -> void* {
    void* p = ws + off;
    off += (bytes + 255) & ~(size_t)255;
    return p;
  };
  h16*   actA  = (h16*)  alloc((size_t)B_ * HW_ * C_ * 2);
  h16*   actB  = (h16*)  alloc((size_t)B_ * HW_ * C_ * 2);
  float* raw   = (float*)alloc((size_t)B_ * HW_ * C_ * 4);
  h16*   wT    = (h16*)  alloc((size_t)7 * 9 * 64 * 64 * 2);
  float* stats = (float*)alloc(B_ * C_ * 2 * 4);
  float* mrs   = (float*)alloc(B_ * C_ * 2 * 4);
  float* lb    = (float*)alloc(169 * 8 * 2 * 4);

  float* hr    = (float*)d_out;
  float* lossO = hr + (size_t)B_ * HW_;

  // Weight transform (runs each call; deterministic).
  k_wprep<<<dim3((7 * 9 * 64 * 64 + 255) / 256), 256, 0, stream>>>(wsA, wT);

  // Layer 0
  (void)hipMemsetAsync(stats, 0, B_ * C_ * 2 * 4, stream);
  k_conv0<<<dim3(B_ * HW_ / 256), 256, 0, stream>>>(xIn, w0, b0, raw, stats);
  k_finalize<<<dim3(2), 256, 0, stream>>>(stats, mrs);
  k_norm<<<dim3((unsigned)((size_t)B_ * HW_ * C_ / 256)), 256, 0, stream>>>(raw, mrs, actA);

  // 7 WMMA mid layers
  h16* cur = actA;
  h16* nxt = actB;
  size_t smem = (size_t)(9 * 64 * 32 + 10 * 18 * 32) * sizeof(uint32_t) + 128 * sizeof(float);
  for (int l = 0; l < 7; ++l) {
    (void)hipMemsetAsync(stats, 0, B_ * C_ * 2 * 4, stream);
    k_conv<<<dim3(16, 32, 8), 256, smem, stream>>>(cur, wT + (size_t)l * 9 * 64 * 64,
                                                   bsA + l * 64, raw, stats);
    k_finalize<<<dim3(2), 256, 0, stream>>>(stats, mrs);
    k_norm<<<dim3((unsigned)((size_t)B_ * HW_ * C_ / 256)), 256, 0, stream>>>(raw, mrs, nxt);
    h16* t = cur; cur = nxt; nxt = t;
  }

  // Projection + residual -> hr (d_out)
  k_proj<<<dim3(B_ * HW_ / 256), 256, 0, stream>>>(cur, wp, bp, xIn, hr);

  // Shift-search loss
  (void)hipMemsetAsync(lb, 0, 169 * 8 * 2 * 4, stream);
  k_lossacc<<<dim3((NPIX_ + 255) / 256, 8, 169), 256, 0, stream>>>(hr, tgt, lb);
  k_lossfin<<<dim3(1), 64, 0, stream>>>(lb, lossO);

  (void)in_sizes; (void)n_in; (void)out_size; (void)ws_size;
}
